// FastMoEEncoder_29111288332646
// MI455X (gfx1250) — matrix operand order repaired and checked by reading, not compiled, not run on hardware
//
#include <hip/hip_runtime.h>
#include <hip/hip_bf16.h>
#include <stdint.h>

typedef _Float16 half_t;
typedef __attribute__((ext_vector_type(16))) _Float16 v16h;
typedef __attribute__((ext_vector_type(8)))  _Float16 v8h;
typedef __attribute__((ext_vector_type(8)))  float    v8f;

#define T_TOKENS 8192
#define DMODEL   768
#define EMBD     256
#define NEXP     8

// ---------------------------------------------------------------------------
// patchify: x[B,3,256,256] -> flat[T,768] (f32 + f16), (C,P,P)-major per patch
// ---------------------------------------------------------------------------
__global__ void patchify_kernel(const float* __restrict__ x,
                                float* __restrict__ flat,
                                half_t* __restrict__ flath) {
  int t  = blockIdx.x;
  int b  = t >> 8;
  int ph = (t >> 4) & 15;
  int pw = t & 15;
  for (int i = threadIdx.x; i < DMODEL; i += blockDim.x) {
    int c  = i >> 8;
    int py = (i >> 4) & 15;
    int px = i & 15;
    size_t src = ((((size_t)b * 3 + c) * 256) + (ph * 16 + py)) * 256 + (pw * 16 + px);
    float v = x[src];
    flat[(size_t)t * DMODEL + i]  = v;
    flath[(size_t)t * DMODEL + i] = (half_t)v;
  }
}

// ---------------------------------------------------------------------------
// zero small aux buffers (deterministic re-init each call)
// ---------------------------------------------------------------------------
__global__ void zero_aux_kernel(int* __restrict__ ce_i, float* __restrict__ me_part, int nPart) {
  int i = blockIdx.x * blockDim.x + threadIdx.x;
  if (i < 8) ce_i[i] = 0;
  if (i < nPart) me_part[i] = 0.f;
}

// ---------------------------------------------------------------------------
// gate: logits = flat@gw+gb, top-2 softmax -> gate_dense; aux-loss partials.
// ce via integer atomics (deterministic); me via in-block tree reduction
// + per-block partial (summed in fixed order later) -> bitwise deterministic.
// ---------------------------------------------------------------------------
__global__ void gate_kernel(const float* __restrict__ flat,
                            const float* __restrict__ gw,
                            const float* __restrict__ gb,
                            float* __restrict__ gate_dense,
                            int* __restrict__ ce_i,
                            float* __restrict__ me_part) {
  __shared__ float sme[256 * 8];
  int tid = threadIdx.x;
  int t = blockIdx.x * 256 + tid;
  float l[8];
  #pragma unroll
  for (int e = 0; e < 8; ++e) l[e] = gb[e];
  const float* fr = flat + (size_t)t * DMODEL;
  for (int k = 0; k < DMODEL; ++k) {
    float fv = fr[k];
    const float* g = gw + k * 8;
    #pragma unroll
    for (int e = 0; e < 8; ++e) l[e] += fv * g[e];
  }
  int i1 = 0; float v1 = l[0];
  #pragma unroll
  for (int e = 1; e < 8; ++e) if (l[e] > v1) { v1 = l[e]; i1 = e; }
  int i2 = -1; float v2 = -3.4e38f;
  #pragma unroll
  for (int e = 0; e < 8; ++e) if (e != i1 && l[e] > v2) { v2 = l[e]; i2 = e; }
  float e2 = __expf(v2 - v1);
  float s1 = 1.f / (1.f + e2);
  float s2 = e2 / (1.f + e2);
  #pragma unroll
  for (int e = 0; e < 8; ++e)
    gate_dense[(size_t)t * 8 + e] = (e == i1) ? s1 : ((e == i2) ? s2 : 0.f);
  atomicAdd(&ce_i[i1], 1);

  float p[8]; float se = 0.f;
  #pragma unroll
  for (int e = 0; e < 8; ++e) { p[e] = __expf(l[e] - v1); se += p[e]; }
  float inv = 1.f / se;
  #pragma unroll
  for (int e = 0; e < 8; ++e) sme[tid * 8 + e] = p[e] * inv;
  __syncthreads();
  for (int s = 128; s > 0; s >>= 1) {
    if (tid < s)
      #pragma unroll
      for (int e = 0; e < 8; ++e) sme[tid * 8 + e] += sme[(tid + s) * 8 + e];
    __syncthreads();
  }
  if (tid == 0)
    #pragma unroll
    for (int e = 0; e < 8; ++e) me_part[blockIdx.x * 8 + e] = sme[e];
}

// ---------------------------------------------------------------------------
// pack_b: W row-major (N,K) f32 -> packed f16 WMMA B tiles.
// Tile (kt index ks, n-tile nt) at offset ((nt*KT+ks)*512): lane L holds
// column n = nt*16 + (L&15); 16 sequential K halves starting at
// ks*32 + ((L>=16)?16:0)  (contiguous k order, matching the A loads).
// ---------------------------------------------------------------------------
__global__ void pack_b_kernel(const float* __restrict__ W, half_t* __restrict__ out,
                              int K, int N) {
  int idx = blockIdx.x * blockDim.x + threadIdx.x;
  int kt = K >> 5, ntiles = N >> 4;
  int total = kt * ntiles * 32;
  if (idx >= total) return;
  int lane = idx & 31;
  int tile = idx >> 5;
  int ks = tile % kt;
  int nt = tile / kt;
  int n = nt * 16 + (lane & 15);
  int koff = ks * 32 + ((lane >> 4) << 4);
  half_t* dst = out + ((size_t)tile << 9) + lane * 16;
  const float* wr = W + (size_t)n * K + koff;
  #pragma unroll
  for (int i = 0; i < 16; ++i) dst[i] = (half_t)wr[i];
}

// ---------------------------------------------------------------------------
// Generic WMMA GEMM: C[M,N] = A[M,K](f16 row-major) x Bpacked + epilogue.
// One wave per 32x64 C-block: 2 M-tiles x 4 N-tiles of
// v_wmma_f32_16x16x32_f16 (8 accumulator tiles), so each streamed B tile is
// reused by two WMMAs (24 B/lane/WMMA vs 40 at 16x64).
// A layout matches packed B (contiguous k): lane L -> row (L&15); halves are two
// contiguous 16-byte runs at k = kb+hiOff and kb+8+hiOff (hiOff = 16*(L>=16)),
// loaded as two v8h -> exactly 2x global_load_b128 per A fragment.
// Epilogue: +bias, optional f32 accumulate, optional ReLU, f32/f16 stores.
// ---------------------------------------------------------------------------
__global__ void wmma_gemm_kernel(const half_t* __restrict__ A,
                                 const half_t* __restrict__ Bp,
                                 int M, int N, int K,
                                 const float* __restrict__ bias,
                                 int relu,
                                 float* __restrict__ Cf, int accum,
                                 half_t* __restrict__ Ch) {
  const int lane = threadIdx.x & 31;
  const int wave = (blockIdx.x * blockDim.x + threadIdx.x) >> 5;
  const int nGroups = N >> 6;
  const int mBlocks = M >> 5;
  if (wave >= mBlocks * nGroups) return;
  const int mb = wave / nGroups;
  const int ng = wave % nGroups;
  const int m0 = mb << 5;
  const int n0 = ng << 6;
  const int kt = K >> 5;
  const int hiOff = (lane >> 4) << 4;              // A K offset 0 or 16 halves
  const half_t* arow0 = A + (size_t)(m0 + (lane & 15)) * K + hiOff;
  const half_t* arow1 = arow0 + (size_t)16 * K;
  const half_t* bbase = Bp + (((size_t)(ng * 4) * kt) << 9) + (size_t)lane * 16;
  const size_t tstride = (size_t)kt << 9;          // halves per n-tile column

  v8f c[2][4] = {};
  for (int ks = 0; ks < kt; ++ks) {
    const int kb = ks << 5;
    const v8h a0lo = *(const v8h*)(arow0 + kb);
    const v8h a0hi = *(const v8h*)(arow0 + kb + 8);
    const v8h a1lo = *(const v8h*)(arow1 + kb);
    const v8h a1hi = *(const v8h*)(arow1 + kb + 8);
    const v16h a0 = __builtin_shufflevector(a0lo, a0hi,
        0, 1, 2, 3, 4, 5, 6, 7, 8, 9, 10, 11, 12, 13, 14, 15);
    const v16h a1 = __builtin_shufflevector(a1lo, a1hi,
        0, 1, 2, 3, 4, 5, 6, 7, 8, 9, 10, 11, 12, 13, 14, 15);
    const half_t* bk = bbase + ((size_t)ks << 9);
    #pragma unroll
    for (int u = 0; u < 4; ++u) {
      const v16h b = *(const v16h*)(bk + (size_t)u * tstride);
      c[0][u] = __builtin_amdgcn_wmma_f32_16x16x32_f16(false, a0, false, b,
                                                       (short)0, c[0][u], false, false);
      c[1][u] = __builtin_amdgcn_wmma_f32_16x16x32_f16(false, a1, false, b,
                                                       (short)0, c[1][u], false, false);
    }
  }

  const int nc = lane & 15;
  #pragma unroll
  for (int h = 0; h < 2; ++h) {
    const int mbase = m0 + h * 16 + ((lane >> 4) << 3);
    #pragma unroll
    for (int u = 0; u < 4; ++u) {
      int n = n0 + u * 16 + nc;
      float bv = bias ? bias[n] : 0.f;
      #pragma unroll
      for (int r = 0; r < 8; ++r) {
        size_t off = (size_t)(mbase + r) * N + n;
        float v = c[h][u][r] + bv;
        if (accum) v += Cf[off];
        if (relu && v < 0.f) v = 0.f;
        if (Cf) Cf[off] = v;
        if (Ch) Ch[off] = (half_t)v;
      }
    }
  }
}

// ---------------------------------------------------------------------------
// Fused dw1(3,s2)+relu -> pw1(3->64)+relu -> dw2(64,s2)+relu per token (LDS),
// emitting A2[t*16+spatial, 64ch] f16 for the pw2 WMMA GEMM.
// ---------------------------------------------------------------------------
__global__ void conv_stage1_kernel(const float* __restrict__ flat,
                                   const float* __restrict__ dw1w,  // 3*9
                                   const float* __restrict__ dw1b,  // 3
                                   const float* __restrict__ pw1w,  // 64*3
                                   const float* __restrict__ pw1b,  // 64
                                   const float* __restrict__ dw2w,  // 64*9
                                   const float* __restrict__ dw2b,  // 64
                                   half_t* __restrict__ A2) {
  __shared__ float p[768];
  __shared__ float d1[192];
  __shared__ float h1[4096];
  __shared__ float w1[27], b1[3], wp1[192], bp1[64], w2[576], b2[64];
  const int t = blockIdx.x;
  const int tid = threadIdx.x;
  for (int i = tid; i < 768; i += 256) p[i] = flat[(size_t)t * 768 + i];
  if (tid < 27)  w1[tid]  = dw1w[tid];
  if (tid < 3)   b1[tid]  = dw1b[tid];
  if (tid < 192) wp1[tid] = pw1w[tid];
  if (tid < 64)  bp1[tid] = pw1b[tid];
  for (int i = tid; i < 576; i += 256) w2[i] = dw2w[i];
  if (tid < 64)  b2[tid]  = dw2b[tid];
  __syncthreads();
  // depthwise 3x3 stride2 pad1: 3ch 16x16 -> 8x8
  if (tid < 192) {
    int c = tid >> 6, s = tid & 63, oy = s >> 3, ox = s & 7;
    float acc = b1[c];
    #pragma unroll
    for (int ky = 0; ky < 3; ++ky) {
      int iy = 2 * oy - 1 + ky; if (iy < 0 || iy > 15) continue;
      #pragma unroll
      for (int kx = 0; kx < 3; ++kx) {
        int ix = 2 * ox - 1 + kx; if (ix < 0 || ix > 15) continue;
        acc += p[c * 256 + iy * 16 + ix] * w1[c * 9 + ky * 3 + kx];
      }
    }
    d1[tid] = acc > 0.f ? acc : 0.f;
  }
  __syncthreads();
  // pointwise 3->64 + relu
  for (int i = tid; i < 4096; i += 256) {
    int oc = i >> 6, s = i & 63;
    float acc = bp1[oc] + wp1[oc * 3 + 0] * d1[s]
                        + wp1[oc * 3 + 1] * d1[64 + s]
                        + wp1[oc * 3 + 2] * d1[128 + s];
    h1[oc * 64 + s] = acc > 0.f ? acc : 0.f;
  }
  __syncthreads();
  // depthwise 3x3 stride2 pad1: 64ch 8x8 -> 4x4, relu, emit A2 [spatial, ch]
  for (int i = tid; i < 1024; i += 256) {
    int c = i >> 4, s = i & 15, oy = s >> 2, ox = s & 3;
    float acc = b2[c];
    #pragma unroll
    for (int ky = 0; ky < 3; ++ky) {
      int iy = 2 * oy - 1 + ky; if (iy < 0 || iy > 7) continue;
      #pragma unroll
      for (int kx = 0; kx < 3; ++kx) {
        int ix = 2 * ox - 1 + kx; if (ix < 0 || ix > 7) continue;
        acc += h1[c * 64 + iy * 8 + ix] * w2[c * 9 + ky * 3 + kx];
      }
    }
    acc = acc > 0.f ? acc : 0.f;
    A2[((size_t)t * 16 + s) * 64 + c] = (half_t)acc;
  }
}

// ---------------------------------------------------------------------------
// dw3: depthwise 3x3 stride2 pad1 on [128ch,4x4] -> [128ch,2x2], relu,
// emits A3[t*4+spatial, 128ch] f16 for pw3 WMMA GEMM.
// ---------------------------------------------------------------------------
__global__ void dw3_kernel(const half_t* __restrict__ H2,   // [T*16,128]
                           const float* __restrict__ w,     // 128*9
                           const float* __restrict__ b,     // 128
                           half_t* __restrict__ A3) {       // [T*4,128]
  int idx = blockIdx.x * blockDim.x + threadIdx.x;
  if (idx >= T_TOKENS * 128) return;
  int t = idx >> 7;
  int c = idx & 127;
  float in[16];
  const half_t* hb = H2 + (size_t)t * 16 * 128 + c;
  #pragma unroll
  for (int s = 0; s < 16; ++s) in[s] = (float)hb[(size_t)s * 128];
  const float* wc = w + c * 9;
  float bc = b[c];
  #pragma unroll
  for (int oy = 0; oy < 2; ++oy)
    #pragma unroll
    for (int ox = 0; ox < 2; ++ox) {
      float acc = bc;
      #pragma unroll
      for (int ky = 0; ky < 3; ++ky) {
        int iy = 2 * oy - 1 + ky; if (iy < 0 || iy > 3) continue;
        #pragma unroll
        for (int kx = 0; kx < 3; ++kx) {
          int ix = 2 * ox - 1 + kx; if (ix < 0 || ix > 3) continue;
          acc += in[iy * 4 + ix] * wc[ky * 3 + kx];
        }
      }
      acc = acc > 0.f ? acc : 0.f;
      A3[((size_t)t * 4 + oy * 2 + ox) * 128 + c] = (half_t)acc;
    }
}

// ---------------------------------------------------------------------------
// AdaptiveAvgPool2d(1) over 2x2 -> feats[T,256] f16
// ---------------------------------------------------------------------------
__global__ void pool_kernel(const half_t* __restrict__ P3, half_t* __restrict__ feats) {
  int idx = blockIdx.x * blockDim.x + threadIdx.x;
  if (idx >= T_TOKENS * 256) return;
  int t = idx >> 8, c = idx & 255;
  const half_t* pb = P3 + (size_t)t * 4 * 256 + c;
  float f = 0.25f * ((float)pb[0] + (float)pb[256] + (float)pb[512] + (float)pb[768]);
  feats[idx] = (half_t)f;
}

// ---------------------------------------------------------------------------
// moe_out (+)= gate[:,e] * expert_out    (e==0 initializes)
// ---------------------------------------------------------------------------
__global__ void combine_kernel(const float* __restrict__ eo,
                               const float* __restrict__ gate,
                               float* __restrict__ moe, int e) {
  int idx = blockIdx.x * blockDim.x + threadIdx.x;
  if (idx >= T_TOKENS * DMODEL) return;
  int t = idx / DMODEL;
  float v = gate[(size_t)t * 8 + e] * eo[idx];
  moe[idx] = (e == 0) ? v : (moe[idx] + v);
}

__global__ void f32_to_f16_kernel(const float* __restrict__ src,
                                  half_t* __restrict__ dst, int n) {
  int idx = blockIdx.x * blockDim.x + threadIdx.x;
  if (idx < n) dst[idx] = (half_t)src[idx];
}

// emb[T,256] -> out[B,256,16,16]
__global__ void output_kernel(const float* __restrict__ emb, float* __restrict__ out) {
  int idx = blockIdx.x * blockDim.x + threadIdx.x;
  if (idx >= T_TOKENS * EMBD) return;
  int t = idx >> 8, c = idx & 255;
  int b = t >> 8, ph = (t >> 4) & 15, pw = t & 15;
  out[(((size_t)b * 256 + c) * 16 + ph) * 16 + pw] = emb[idx];
}

// l_aux = E * sum_e( (ce_e/T) * (me_e/T) ), summed in fixed order
__global__ void laux_kernel(const int* __restrict__ ce_i,
                            const float* __restrict__ me_part,
                            int nBlocks, float* __restrict__ out_scalar) {
  if (threadIdx.x != 0 || blockIdx.x != 0) return;
  float acc = 0.f;
  for (int e = 0; e < 8; ++e) {
    float me = 0.f;
    for (int p = 0; p < nBlocks; ++p) me += me_part[p * 8 + e];
    acc += ((float)ce_i[e] / (float)T_TOKENS) * (me / (float)T_TOKENS);
  }
  *out_scalar = 8.f * acc;
}

// ---------------------------------------------------------------------------
extern "C" void kernel_launch(void* const* d_in, const int* in_sizes, int n_in,
                              void* d_out, int out_size, void* d_ws, size_t ws_size,
                              hipStream_t stream) {
  (void)in_sizes; (void)n_in; (void)out_size; (void)ws_size;
  const float* x    = (const float*)d_in[0];
  const float* gw   = (const float*)d_in[1];
  const float* gb   = (const float*)d_in[2];
  const float* dw1w = (const float*)d_in[3];
  const float* dw1b = (const float*)d_in[4];
  const float* pw1w = (const float*)d_in[5];
  const float* pw1b = (const float*)d_in[6];
  const float* dw2w = (const float*)d_in[7];
  const float* dw2b = (const float*)d_in[8];
  const float* pw2w = (const float*)d_in[9];
  const float* pw2b = (const float*)d_in[10];
  const float* dw3w = (const float*)d_in[11];
  const float* dw3b = (const float*)d_in[12];
  const float* pw3w = (const float*)d_in[13];
  const float* pw3b = (const float*)d_in[14];
  const float* thw  = (const float*)d_in[15];
  const float* thb  = (const float*)d_in[16];
  const float* resw = (const float*)d_in[17];
  const float* resb = (const float*)d_in[18];
  const float* tew  = (const float*)d_in[19];
  const float* teb  = (const float*)d_in[20];
  float* out = (float*)d_out;

  // ---- workspace carve (256B aligned) ----
  uint8_t* wp = (uint8_t*)d_ws;
  auto carve = [&](size_t bytes) -> void* {
    void* p = (void*)wp; wp += (bytes + 255) & ~(size_t)255; return p;
  };
  const size_t T = T_TOKENS;
  float*  flatf  = (float*) carve(T * DMODEL * 4);
  half_t* flath  = (half_t*)carve(T * DMODEL * 2);
  half_t* resBp  = (half_t*)carve((size_t)NEXP * 768 * 768 * 2);
  half_t* thBp   = (half_t*)carve((size_t)NEXP * 256 * 768 * 2);
  half_t* teBp   = (half_t*)carve((size_t)768 * 256 * 2);
  half_t* pw2Bp  = (half_t*)carve((size_t)NEXP * 64 * 128 * 2);
  half_t* pw3Bp  = (half_t*)carve((size_t)NEXP * 128 * 256 * 2);
  float*  gateD  = (float*) carve(T * 8 * 4);
  int*    ce_i   = (int*)   carve(8 * 4);
  float*  mePart = (float*) carve(32 * 8 * 4);
  half_t* A2     = (half_t*)carve(T * 16 * 64 * 2);
  half_t* H2h    = (half_t*)carve(T * 16 * 128 * 2);
  half_t* A3     = (half_t*)carve(T * 4 * 128 * 2);
  half_t* P3     = (half_t*)carve(T * 4 * 256 * 2);
  half_t* feats  = (half_t*)carve(T * 256 * 2);
  float*  eout   = (float*) carve(T * DMODEL * 4);
  float*  moe    = (float*) carve(T * DMODEL * 4);
  half_t* moeh   = (half_t*)carve(T * DMODEL * 2);
  float*  embf   = (float*) carve(T * EMBD * 4);

  auto gemm = [&](const half_t* A, const half_t* Bp, int M, int N, int K,
                  const float* bias, int relu, float* Cf, int accum, half_t* Ch) {
    int waves = (M >> 5) * (N >> 6);          // 32x64 C-block per wave
    int blocks = (waves + 7) / 8;
    wmma_gemm_kernel<<<blocks, 256, 0, stream>>>(A, Bp, M, N, K, bias, relu, Cf, accum, Ch);
  };
  auto pack = [&](const float* W, half_t* dst, int K, int N) {
    int total = (K >> 5) * (N >> 4) * 32;
    pack_b_kernel<<<(total + 255) / 256, 256, 0, stream>>>(W, dst, K, N);
  };

  // 1) patchify + f16 copy
  patchify_kernel<<<T_TOKENS, 256, 0, stream>>>(x, flatf, flath);

  // 2) pack all weight matrices into WMMA-B layout (f16)
  for (int e = 0; e < NEXP; ++e) {
    pack(resw + (size_t)e * 768 * 768, resBp + (size_t)e * 768 * 768, 768, 768);
    pack(thw  + (size_t)e * 768 * 256, thBp  + (size_t)e * 256 * 768, 256, 768);
    pack(pw2w + (size_t)e * 128 * 64,  pw2Bp + (size_t)e * 64 * 128,  64,  128);
    pack(pw3w + (size_t)e * 256 * 128, pw3Bp + (size_t)e * 128 * 256, 128, 256);
  }
  pack(tew, teBp, 768, 256);

  // 3) gate + aux partials
  zero_aux_kernel<<<1, 256, 0, stream>>>(ce_i, mePart, 256);
  gate_kernel<<<32, 256, 0, stream>>>(flatf, gw, gb, gateD, ce_i, mePart);

  // 4) dense expert loop
  for (int e = 0; e < NEXP; ++e) {
    conv_stage1_kernel<<<T_TOKENS, 256, 0, stream>>>(
        flatf, dw1w + e * 27, dw1b + e * 3, pw1w + e * 192, pw1b + e * 64,
        dw2w + e * 576, dw2b + e * 64, A2);
    gemm(A2, pw2Bp + (size_t)e * 64 * 128, T_TOKENS * 16, 128, 64,
         pw2b + e * 128, 1, nullptr, 0, H2h);
    dw3_kernel<<<(T_TOKENS * 128 + 255) / 256, 256, 0, stream>>>(
        H2h, dw3w + e * 1152, dw3b + e * 128, A3);
    gemm(A3, pw3Bp + (size_t)e * 128 * 256, T_TOKENS * 4, 256, 128,
         pw3b + e * 256, 1, nullptr, 0, P3);
    pool_kernel<<<(T_TOKENS * 256 + 255) / 256, 256, 0, stream>>>(P3, feats);
    // expert_out = feats @ th^T + th_b ; += flat @ res^T + res_b
    gemm(feats, thBp + (size_t)e * 256 * 768, T_TOKENS, 768, 256,
         thb + e * 768, 0, eout, 0, nullptr);
    gemm(flath, resBp + (size_t)e * 768 * 768, T_TOKENS, 768, 768,
         resb + e * 768, 0, eout, 1, nullptr);
    combine_kernel<<<(T_TOKENS * DMODEL + 255) / 256, 256, 0, stream>>>(
        eout, gateD, moe, e);
  }

  // 5) to_emb GEMM + output permute + aux loss
  f32_to_f16_kernel<<<(T_TOKENS * DMODEL + 255) / 256, 256, 0, stream>>>(
      moe, moeh, T_TOKENS * DMODEL);
  gemm(moeh, teBp, T_TOKENS, 256, 768, teb, 0, embf, 0, nullptr);
  output_kernel<<<(T_TOKENS * EMBD + 255) / 256, 256, 0, stream>>>(embf, out);
  laux_kernel<<<1, 32, 0, stream>>>(ce_i, mePart, 32, out + (size_t)T_TOKENS * EMBD);
}
